// MultiHeadAttention_19043884990948
// MI455X (gfx1250) — matrix operand order repaired and checked by reading, not compiled, not run on hardware
//
#include <hip/hip_runtime.h>

// ---------------------------------------------------------------------------
// MHA forward for MI455X (gfx1250, wave32, WMMA bf16).
// B=2, S=2048, D=1024, H=16, HD=64.
//
//   0) fp32->bf16 conversion of hidden; weights converted+transposed K-major
//      (Wq pre-scaled by 1/sqrt(HD) so attention needs no score scaling).
//   1) QKV projection GEMMs (32x64 tile / wave), V stored transposed [HD,S].
//   2) flash attention, 32 query rows / wave; K blocks streamed into a
//      double-buffered LDS tile with global_load_async_to_lds_b128
//      (ASYNCcnt pipeline), fp32 online softmax with DPP16 butterfly
//      reductions, P re-laid out through LDS (s_wait_dscnt).
//   3) output projection GEMM + bias, fp32 out.
// ---------------------------------------------------------------------------

#define BATCH 2
#define SEQ   2048
#define DIM   1024
#define NH    16
#define HDIM  64

typedef __attribute__((ext_vector_type(16))) __bf16 bfx16;
typedef __attribute__((ext_vector_type(8)))  __bf16 bfx8;
typedef __attribute__((ext_vector_type(4)))  __bf16 bfx4;
typedef __attribute__((ext_vector_type(8)))  float  f32x8;

static __device__ __forceinline__ f32x8 wmma_bf16(bfx16 a, bfx16 b, f32x8 c) {
  // (neg_a, A, neg_b, B, c_mod, C, reuse_a, reuse_b)
  return __builtin_amdgcn_wmma_f32_16x16x32_bf16(false, a, false, b, (short)0, c,
                                                 false, false);
}

static __device__ __forceinline__ bfx16 concat8(bfx8 lo, bfx8 hi) {
  return __builtin_shufflevector(lo, hi, 0, 1, 2, 3, 4, 5, 6, 7,
                                 8, 9, 10, 11, 12, 13, 14, 15);
}

// A-fragment (16x32, MxK) from bf16 row-major (row stride lda elements).
// Lane layout: element i -> k = (i<8 ? i : i+8) + 8*hi  => two 16B runs.
static __device__ __forceinline__ bfx16 load_a_bf(const __bf16* __restrict__ p,
                                                  int lda, int m, int hi) {
  const __bf16* row = p + (size_t)m * lda + hi * 8;
  return concat8(*(const bfx8*)row, *(const bfx8*)(row + 16));
}

// B-fragment (32x16, KxN) from a K-contiguous layout: element(k,n) = p[n*ld+k].
// Lane layout: element i -> k = (i<8 ? i : i+8) + 8*hi  => two 16B runs.
static __device__ __forceinline__ bfx16 load_bk_bf(const __bf16* __restrict__ p,
                                                   int ld, int n, int hi) {
  const __bf16* col = p + (size_t)n * ld + hi * 8;
  return concat8(*(const bfx8*)col, *(const bfx8*)(col + 16));
}

// ---- DPP16 butterfly reduction within each 16-lane half (no LDS traffic) ----
// quad_perm{1,0,3,2}=0xB1, quad_perm{2,3,0,1}=0x4E, row_half_mirror=0x141,
// row_mirror=0x140.  All stay inside a 16-lane row on wave32.
#define DPP_F32(x, ctrl)                                                  \
  __int_as_float(__builtin_amdgcn_update_dpp(                             \
      __float_as_int(x), __float_as_int(x), (ctrl), 0xF, 0xF, true))

static __device__ __forceinline__ float red_max16(float x) {
  x = fmaxf(x, DPP_F32(x, 0xB1));
  x = fmaxf(x, DPP_F32(x, 0x4E));
  x = fmaxf(x, DPP_F32(x, 0x141));
  x = fmaxf(x, DPP_F32(x, 0x140));
  return x;
}
static __device__ __forceinline__ float red_sum16(float x) {
  x += DPP_F32(x, 0xB1);
  x += DPP_F32(x, 0x4E);
  x += DPP_F32(x, 0x141);
  x += DPP_F32(x, 0x140);
  return x;
}

// ---- async global->LDS copy of one contiguous 4KB block (8 x b128/lane) ----
static __device__ __forceinline__ void async_copy_4k(unsigned lds_byte,
                                                     const __bf16* gptr,
                                                     int lane) {
  unsigned long long ga = (unsigned long long)(uintptr_t)gptr +
                          (unsigned)(lane * 16);
  unsigned la = lds_byte + (unsigned)(lane * 16);
  asm volatile(
      "global_load_async_to_lds_b128 %0, %1, off\n\t"
      "global_load_async_to_lds_b128 %0, %1, off offset:512\n\t"
      "global_load_async_to_lds_b128 %0, %1, off offset:1024\n\t"
      "global_load_async_to_lds_b128 %0, %1, off offset:1536\n\t"
      "global_load_async_to_lds_b128 %0, %1, off offset:2048\n\t"
      "global_load_async_to_lds_b128 %0, %1, off offset:2560\n\t"
      "global_load_async_to_lds_b128 %0, %1, off offset:3072\n\t"
      "global_load_async_to_lds_b128 %0, %1, off offset:3584"
      :: "v"(la), "v"(ga) : "memory");
}

// ---------------------------------------------------------------------------
// Prologue 0a: hidden fp32 -> bf16 (element count multiple of 4).
// ---------------------------------------------------------------------------
__global__ __launch_bounds__(256) void cvt_f32_bf16_kernel(
    const float* __restrict__ in, __bf16* __restrict__ out) {
  size_t i = ((size_t)blockIdx.x * blockDim.x + threadIdx.x) * 4;
  float4 v = *(const float4*)(in + i);
  bfx4 o;
  o[0] = (__bf16)v.x; o[1] = (__bf16)v.y; o[2] = (__bf16)v.z; o[3] = (__bf16)v.w;
  *(bfx4*)(out + i) = o;
}

// ---------------------------------------------------------------------------
// Prologue 0b: transpose+convert(+scale) weights: [z,rows,cols] fp32 ->
//              [z,cols,rows] bf16 (K-major for B-fragment loads).
// ---------------------------------------------------------------------------
__global__ __launch_bounds__(256) void transpose_w_kernel(
    const float* __restrict__ in, __bf16* __restrict__ out,
    int rows, int cols, float scale) {
  int r = blockIdx.x * blockDim.x + threadIdx.x;  // coalesced on output
  int c = blockIdx.y;
  int z = blockIdx.z;
  out[((size_t)z * cols + c) * rows + r] =
      (__bf16)(in[((size_t)z * rows + r) * cols + c] * scale);
}

// ---------------------------------------------------------------------------
// Phase 1: per-head projection.  One wave -> 32x64 output tile.
//   X  : [B*S, D]  bf16;  Wt : [H, HD, D] bf16 (K-major)
//   out: VT ? [B,H,HD,S] (for V) : [B,H,S,HD] (for Q,K)
// ---------------------------------------------------------------------------
template <bool VT>
__global__ __launch_bounds__(32) void proj_kernel(
    const __bf16* __restrict__ X, const __bf16* __restrict__ Wt,
    const float* __restrict__ bias, float bscale, __bf16* __restrict__ out) {
  const int lane = threadIdx.x;
  const int m = lane & 15, hi = lane >> 4;
  const int sblk = blockIdx.x;  // S/32
  const int bh = blockIdx.y;    // B*H
  const int h = bh % NH;

  const __bf16* Abase = X + ((size_t)(bh / NH) * SEQ + sblk * 32) * DIM;
  const __bf16* Bbase = Wt + (size_t)h * HDIM * DIM;

  f32x8 acc[2][4] = {};
  for (int k0 = 0; k0 < DIM; k0 += 32) {
    bfx16 a0 = load_a_bf(Abase + k0, DIM, m, hi);
    bfx16 a1 = load_a_bf(Abase + (size_t)16 * DIM + k0, DIM, m, hi);
#pragma unroll
    for (int nt = 0; nt < 4; ++nt) {
      bfx16 b = load_bk_bf(Bbase + (size_t)nt * 16 * DIM + k0, DIM, m, hi);
      acc[0][nt] = wmma_bf16(a0, b, acc[0][nt]);
      acc[1][nt] = wmma_bf16(a1, b, acc[1][nt]);
    }
  }

#pragma unroll
  for (int mt = 0; mt < 2; ++mt)
#pragma unroll
    for (int nt = 0; nt < 4; ++nt) {
      int n = nt * 16 + m;
      float bi = bias[h * HDIM + n] * bscale;
      if (VT) {
        // transposed store [B,H,HD,S]: each lane writes 8 contiguous bf16.
        bfx8 vv;
#pragma unroll
        for (int r = 0; r < 8; ++r) vv[r] = (__bf16)(acc[mt][nt][r] + bi);
        __bf16* ptr = out + ((size_t)bh * HDIM + n) * SEQ + sblk * 32 +
                      mt * 16 + hi * 8;
        *(bfx8*)ptr = vv;
      } else {
        __bf16* ob = out + ((size_t)bh * SEQ + sblk * 32 + mt * 16) * HDIM;
#pragma unroll
        for (int r = 0; r < 8; ++r)
          ob[(size_t)(r + 8 * hi) * HDIM + n] = (__bf16)(acc[mt][nt][r] + bi);
      }
    }
}

// ---------------------------------------------------------------------------
// Phase 2: flash attention.  One wave owns 32 query rows of one (b,h).
//   Q,K: [B,H,S,HD] bf16 (Q pre-scaled);  Vt: [B,H,HD,S];  ctx: [B,S,D].
// K blocks (contiguous 4KB) are streamed into LDS with the async engine.
// ---------------------------------------------------------------------------
__global__ __launch_bounds__(32) void attn_kernel(
    const __bf16* __restrict__ Q, const __bf16* __restrict__ K,
    const __bf16* __restrict__ Vt, __bf16* __restrict__ ctx) {
  __shared__ __bf16 k_lds[2][32 * HDIM];  // 2 x 4KB K double buffer
  __shared__ __bf16 p_lds[32 * 32];       // 2KB P re-layout buffer

  const int lane = threadIdx.x;
  const int m = lane & 15, hi = lane >> 4;
  const int sblk = blockIdx.x;  // S/32
  const int bh = blockIdx.y;

  const __bf16* Qb = Q + ((size_t)bh * SEQ + sblk * 32) * HDIM;
  const __bf16* Kb = K + (size_t)bh * SEQ * HDIM;
  const __bf16* Vb = Vt + (size_t)bh * HDIM * SEQ;

  const unsigned klds_byte[2] = {
      (unsigned)(uintptr_t)&k_lds[0][0],   // flat low 32 bits == LDS offset
      (unsigned)(uintptr_t)&k_lds[1][0]};

  // Q tile 32x64 as 2x2 A-fragments, loaded once (already has 1/sqrt(HD)).
  bfx16 qa[2][2];
#pragma unroll
  for (int mt = 0; mt < 2; ++mt)
#pragma unroll
    for (int t = 0; t < 2; ++t)
      qa[mt][t] = load_a_bf(Qb + (size_t)mt * 16 * HDIM + t * 32, HDIM, m, hi);

  float mrow[2][8], lrow[2][8];
#pragma unroll
  for (int mt = 0; mt < 2; ++mt)
#pragma unroll
    for (int r = 0; r < 8; ++r) { mrow[mt][r] = -3.0e38f; lrow[mt][r] = 0.0f; }
  f32x8 acc[2][4] = {};

  // Preload K block 0 into LDS buffer 0.
  async_copy_4k(klds_byte[0], Kb, lane);

  for (int kb = 0; kb < SEQ; kb += 32) {
    const int cur = (kb >> 5) & 1;
    const bool more = (kb + 32) < SEQ;

    if (more) {
      // prior DS reads of buf[cur^1] must be done before DMA overwrites it
      asm volatile("s_wait_dscnt 0x0" ::: "memory");
      async_copy_4k(klds_byte[cur ^ 1], Kb + (size_t)(kb + 32) * HDIM, lane);
      __builtin_prefetch(Vb + (size_t)lane * SEQ + kb + 32, 0, 1);
      __builtin_prefetch(Vb + (size_t)(32 + lane) * SEQ + kb + 32, 0, 1);
      asm volatile("s_wait_asynccnt 0x8" ::: "memory");  // buf[cur] ready
    } else {
      asm volatile("s_wait_asynccnt 0x0" ::: "memory");
    }

    // ---- scores S = Q K^T : 2x2 C tiles; K^T B-frags from LDS ----
    const __bf16* kbuf = &k_lds[cur][0];
    bfx16 bt[2][2];
#pragma unroll
    for (int kt = 0; kt < 2; ++kt)
#pragma unroll
      for (int t = 0; t < 2; ++t)
        bt[kt][t] = load_bk_bf(kbuf + (size_t)kt * 16 * HDIM + t * 32,
                               HDIM, m, hi);
    f32x8 s[2][2] = {};
#pragma unroll
    for (int mt = 0; mt < 2; ++mt)
#pragma unroll
      for (int kt = 0; kt < 2; ++kt)
#pragma unroll
        for (int t = 0; t < 2; ++t)
          s[mt][kt] = wmma_bf16(qa[mt][t], bt[kt][t], s[mt][kt]);

    // ---- online softmax per 16-row half (DPP reductions, no LDS) ----
#pragma unroll
    for (int mt = 0; mt < 2; ++mt) {
      float rsum[8], corr[8];
#pragma unroll
      for (int r = 0; r < 8; ++r) {
        float tmax = red_max16(fmaxf(s[mt][0][r], s[mt][1][r]));
        float mn = fmaxf(mrow[mt][r], tmax);
        corr[r] = __expf(mrow[mt][r] - mn);
        mrow[mt][r] = mn;
        float p0 = __expf(s[mt][0][r] - mn);
        float p1 = __expf(s[mt][1][r] - mn);
        rsum[r] = red_sum16(p0 + p1);
        int row = mt * 16 + r + 8 * hi;
        p_lds[row * 32 + m] = (__bf16)p0;
        p_lds[row * 32 + 16 + m] = (__bf16)p1;
      }
#pragma unroll
      for (int r = 0; r < 8; ++r) {
        lrow[mt][r] = lrow[mt][r] * corr[r] + rsum[r];
#pragma unroll
        for (int nt = 0; nt < 4; ++nt) acc[mt][nt][r] *= corr[r];
      }
    }

    // cross-lane LDS dependency within the wave: drain DS counter.
    asm volatile("s_wait_dscnt 0x0" ::: "memory");
    __builtin_amdgcn_wave_barrier();

    // ---- acc += P @ V : P from LDS as A-frags, V-frags reused across mt ----
    bfx16 pa[2];
#pragma unroll
    for (int mt = 0; mt < 2; ++mt) {
      const __bf16* prow = &p_lds[(mt * 16 + m) * 32 + hi * 8];
      pa[mt] = concat8(*(const bfx8*)prow, *(const bfx8*)(prow + 16));
    }
#pragma unroll
    for (int nt = 0; nt < 4; ++nt) {
      bfx16 vb = load_bk_bf(Vb + (size_t)nt * 16 * SEQ + kb, SEQ, m, hi);
      acc[0][nt] = wmma_bf16(pa[0], vb, acc[0][nt]);
      acc[1][nt] = wmma_bf16(pa[1], vb, acc[1][nt]);
    }
    __builtin_amdgcn_wave_barrier();  // keep next DS writes behind these reads
  }

  // ---- normalize and store ctx in concat-heads layout [B,S,D] ----
  const int b = bh / NH, h = bh % NH;
  __bf16* ob = ctx + (((size_t)b * SEQ + sblk * 32) * NH + h) * HDIM;
#pragma unroll
  for (int mt = 0; mt < 2; ++mt)
#pragma unroll
    for (int nt = 0; nt < 4; ++nt) {
      int e = nt * 16 + m;
#pragma unroll
      for (int r = 0; r < 8; ++r) {
        int row = mt * 16 + r + 8 * hi;
        ob[(size_t)row * DIM + e] = (__bf16)(acc[mt][nt][r] / lrow[mt][r]);
      }
    }
}

// ---------------------------------------------------------------------------
// Phase 3: out = ctx @ Wo + bo.  One wave -> 32x64 tile, Wo^T is K-major.
// ---------------------------------------------------------------------------
__global__ __launch_bounds__(32) void out_proj_kernel(
    const __bf16* __restrict__ ctx,   // [M, D] bf16
    const __bf16* __restrict__ Wot,   // [D, D] bf16, Wot[n*D + k] = Wo[k, n]
    const float* __restrict__ bo,     // [D]
    float* __restrict__ out) {        // [M, D] fp32
  const int lane = threadIdx.x;
  const int m = lane & 15, hi = lane >> 4;
  const int mblk = blockIdx.x;  // M/32
  const int nblk = blockIdx.y;  // D/64

  const __bf16* Abase = ctx + (size_t)mblk * 32 * DIM;
  f32x8 acc[2][4] = {};
  for (int k0 = 0; k0 < DIM; k0 += 32) {
    bfx16 a0 = load_a_bf(Abase + k0, DIM, m, hi);
    bfx16 a1 = load_a_bf(Abase + (size_t)16 * DIM + k0, DIM, m, hi);
#pragma unroll
    for (int nt = 0; nt < 4; ++nt) {
      bfx16 b = load_bk_bf(Wot + (size_t)(nblk * 64 + nt * 16) * DIM + k0,
                           DIM, m, hi);
      acc[0][nt] = wmma_bf16(a0, b, acc[0][nt]);
      acc[1][nt] = wmma_bf16(a1, b, acc[1][nt]);
    }
  }

  float* ob = out + (size_t)mblk * 32 * DIM + nblk * 64;
#pragma unroll
  for (int mt = 0; mt < 2; ++mt)
#pragma unroll
    for (int nt = 0; nt < 4; ++nt) {
      int n = nt * 16 + m;
      float bi = bo[nblk * 64 + n];
#pragma unroll
      for (int r = 0; r < 8; ++r)
        ob[(size_t)(mt * 16 + r + 8 * hi) * DIM + n] = acc[mt][nt][r] + bi;
    }
}

// ---------------------------------------------------------------------------
extern "C" void kernel_launch(void* const* d_in, const int* in_sizes, int n_in,
                              void* d_out, int out_size, void* d_ws,
                              size_t ws_size, hipStream_t stream) {
  (void)in_sizes; (void)n_in; (void)out_size; (void)ws_size;

  const float* hidden = (const float*)d_in[0];
  const float* Wq = (const float*)d_in[1];
  const float* bq = (const float*)d_in[2];
  const float* Wk = (const float*)d_in[3];
  const float* bk = (const float*)d_in[4];
  const float* Wv = (const float*)d_in[5];
  const float* bv = (const float*)d_in[6];
  const float* Wo = (const float*)d_in[7];
  const float* bo = (const float*)d_in[8];
  float* out = (float*)d_out;

  const float qscale = 0.125f;  // 1/sqrt(HD), folded into Wq/bq
  const size_t nX = (size_t)BATCH * SEQ * DIM;        // 4M elems
  const size_t nW = (size_t)NH * DIM * HDIM;          // 1M elems
  __bf16* x_bf  = (__bf16*)d_ws;                      // [B*S, D]
  __bf16* wq_t  = x_bf + nX;                          // [H, HD, D]
  __bf16* wk_t  = wq_t + nW;
  __bf16* wv_t  = wk_t + nW;
  __bf16* wo_t  = wv_t + nW;                          // [D, D] (N-major)
  __bf16* q_ws  = wo_t + (size_t)DIM * DIM;           // [B,H,S,HD]
  __bf16* k_ws  = q_ws + nX;
  __bf16* vt_ws = k_ws + nX;                          // [B,H,HD,S]
  __bf16* c_ws  = vt_ws + nX;                         // [B,S,D]

  // Phase 0: convert + transpose (+ fold attention scale into Wq).
  cvt_f32_bf16_kernel<<<nX / 4 / 256, 256, 0, stream>>>(hidden, x_bf);
  {
    dim3 tg(DIM / 256, HDIM, NH);
    transpose_w_kernel<<<tg, 256, 0, stream>>>(Wq, wq_t, DIM, HDIM, qscale);
    transpose_w_kernel<<<tg, 256, 0, stream>>>(Wk, wk_t, DIM, HDIM, 1.0f);
    transpose_w_kernel<<<tg, 256, 0, stream>>>(Wv, wv_t, DIM, HDIM, 1.0f);
    transpose_w_kernel<<<dim3(DIM / 256, DIM, 1), 256, 0, stream>>>(
        Wo, wo_t, DIM, DIM, 1.0f);
  }

  // Phase 1: projections (V transposed for K-major PV fragments).
  dim3 pg(SEQ / 32, BATCH * NH);
  proj_kernel<false><<<pg, 32, 0, stream>>>(x_bf, wq_t, bq, qscale, q_ws);
  proj_kernel<false><<<pg, 32, 0, stream>>>(x_bf, wk_t, bk, 1.0f, k_ws);
  proj_kernel<true ><<<pg, 32, 0, stream>>>(x_bf, wv_t, bv, 1.0f, vt_ws);

  // Phase 2: flash attention (async-LDS K streaming).
  attn_kernel<<<dim3(SEQ / 32, BATCH * NH), 32, 0, stream>>>(q_ws, k_ws, vt_ws,
                                                             c_ws);

  // Phase 3: output projection.
  out_proj_kernel<<<dim3(BATCH * SEQ / 32, DIM / 64), 32, 0, stream>>>(
      c_ws, wo_t, bo, out);
}
